// FusedNeRF_46471546142971
// MI455X (gfx1250) — matrix operand order repaired and checked by reading, not compiled, not run on hardware
//
#include <hip/hip_runtime.h>

typedef _Float16 h16;
typedef __attribute__((ext_vector_type(16))) _Float16 v16h;
typedef __attribute__((ext_vector_type(8)))  _Float16 v8h;
typedef __attribute__((ext_vector_type(8)))  float    v8f;
typedef __attribute__((ext_vector_type(4)))  unsigned int u32x4;
typedef __attribute__((ext_vector_type(8)))  int          i32x8;
typedef __attribute__((ext_vector_type(4)))  int          i32x4;

#define XCH 90   // 63 pos + 27 view

// ---------------- Weight image layout (offsets in half elements) ----------------
// All weights transposed: WT[n][k], K padded to mult of 32, f16. Same image in d_ws and LDS.
enum : int {
  OFF_D1W0 = 0,                    // [128][64]   (K real 63)
  OFF_D1W1 = OFF_D1W0 + 128*64,    // [128][128]
  OFF_D1W2 = OFF_D1W1 + 128*128,   // [128][128]
  OFF_D2W0 = OFF_D1W2 + 128*128,   // [128][192]  rows reordered: [part1(128)|pos(63)|pad]
  OFF_D2W1 = OFF_D2W0 + 128*192,   // [128][128]
  OFF_D2W2 = OFF_D2W1 + 128*128,   // [128][128]
  OFF_D2W3 = OFF_D2W2 + 128*128,   // [144][128]  cols reordered: n<128=feature, n=128=density
  OFF_CW0  = OFF_D2W3 + 144*128,   // [128][160]  rows reordered: [feature(128)|view(27)|pad]
  OFF_CW1  = OFF_CW0  + 128*160,   // [16][128]   n<3 = rgb
  LDS_HALFS = OFF_CW1 + 16*128     // 139,264 halfs = 278,528 B (weights only)
};

#define WMMA(a, b, c) __builtin_amdgcn_wmma_f32_16x16x32_f16(false, (a), false, (b), (short)0, (c), false, false)

union TA { v16h v; v8h h[2]; };

// B fragment (32x16 f16): lane L holds col n=L&15; element e: K = (e>>3)*16 + (L>>4)*8 + (e&7).
// Weights transposed [n][k] -> two contiguous 16B runs per lane, 32B apart (two ds_load_b128).
__device__ __forceinline__ v16h load_fragB(const h16* wt, int kstride, int n0, int k0, int lane) {
  const int n = lane & 15, hf = lane >> 4;
  const h16* p = wt + (n0 + n)*kstride + k0 + hf*8;
  union { v16h v; v8h h[2]; } f;
  f.h[0] = *(const v8h*)(p);
  f.h[1] = *(const v8h*)(p + 16);
  return f.v;
}

// Fast A fragment from global x, compile-time masking. Virtual vector v[k] = x[row*90+SRC0+k]
// for k<LIMIT else 0. Element e -> K = k0 + (e>>3)*16 + hf*8 + (e&7). All addresses provably
// in-row: unconditional case reads k <= LIMIT-1; hf-masked case reads p[kb] with kb < LIMIT.
template<int SRC0, int LIMIT>
__device__ __forceinline__ v16h load_x_frag_fast(const float* __restrict__ x, int row, int k0, int hf) {
  union { v16h v; h16 e[16]; } f;
  const float* p = x + (size_t)row * XCH + SRC0;
#pragma unroll
  for (int e = 0; e < 16; ++e) {
    const int kb = k0 + (e >> 3)*16 + (e & 7);      // compile-time after unroll
    float val;
    if (kb + 8 < LIMIT)      val = __builtin_nontemporal_load(p + kb + hf*8);  // both halves valid
    else if (kb < LIMIT)     val = hf ? 0.0f : __builtin_nontemporal_load(p + kb); // hf=0 only
    else                     val = 0.0f;                                       // padding
    f.e[e] = (h16)val;
  }
  return f.v;
}

// Slow fallback (partial tail tile): fully runtime-masked.
__device__ __forceinline__ v16h load_x_frag_slow(const float* __restrict__ x, int row, int nrows,
                                                 int src0, int limit, int k0, int hf) {
  union { v16h v; h16 e[16]; } f;
  const float* p = x + (size_t)row * XCH + src0;
  const bool rv = row < nrows;
#pragma unroll
  for (int e = 0; e < 16; ++e) {
    int k = k0 + (e >> 3)*16 + hf*8 + (e & 7);
    float val = (rv && k < limit) ? __builtin_nontemporal_load(p + k) : 0.0f;
    f.e[e] = (h16)val;
  }
  return f.v;
}

__device__ __forceinline__ v8h cvt_pack(v8f a) {
  v8h r;
#pragma unroll
  for (int i = 0; i < 8; ++i) r[i] = (h16)a[i];   // 4x v_cvt_pk_f16_f32
  return r;
}

// One layer, 32 rows x (KT*32) @ WT -> 32 x 128, with in-register transpose of the output
// into next-layer A fragments (An). Transpose trick: feed cvt-packed D as the A operand vs an
// identity B; by the A layout, lane-half hf holds exactly the k-range (=old rows) it already
// owns, so the operand is [pack(acc) | 0] with no cross-lane moves. The transpose WMMA uses the
// f16-D variant: identity means one nonzero product per output (exact in f16), and the 16-bit
// D layout (element h -> M = h + 8*hf) IS the next layer's A chunk layout -- no conversion.
// ReLU commutes with the transpose, applied afterwards as packed f16 max.
// B fragments are preloaded per n-tile (one dscnt wait per n-tile, uninterrupted WMMA chain);
// ta is double-buffered so consecutive transposes have no WAR hazard on the A operand block.
template<int KT, bool RELU>
__device__ __forceinline__ void layer_T(const v16h (&A)[6][2], const h16* wt, int kw,
                                        v16h (&An)[4][2], v16h idB, TA (&ta)[2], int lane) {
  v8h lo0, lo1;
#pragma unroll
  for (int nt = 0; nt < 8; ++nt) {
    v16h b[KT];
#pragma unroll
    for (int kt = 0; kt < KT; ++kt) b[kt] = load_fragB(wt, kw, nt*16, kt*32, lane);
    v8f acc0 = {}, acc1 = {};
#pragma unroll
    for (int kt = 0; kt < KT; ++kt) {
      acc0 = WMMA(A[kt][0], b[kt], acc0);
      acc1 = WMMA(A[kt][1], b[kt], acc1);
    }
#pragma unroll
    for (int mt = 0; mt < 2; ++mt) {
      ta[mt].h[0] = cvt_pack(mt ? acc1 : acc0);
      v8h zero8 = {};
#if __has_builtin(__builtin_amdgcn_wmma_f16_16x16x32_f16)
      v8h p = __builtin_amdgcn_wmma_f16_16x16x32_f16(false, ta[mt].v, false, idB,
                                                     (short)0, zero8, false, false);
#else
      v8f zero = {};
      v8f d = WMMA(ta[mt].v, idB, zero);
      v8h p = cvt_pack(d);
#endif
      if (RELU) p = __builtin_elementwise_max(p, zero8);   // v_pk_max_num_f16
      if ((nt & 1) == 0) { if (mt) lo1 = p; else lo0 = p; }
      else {
        union { v16h v; v8h h[2]; } u;
        u.h[0] = mt ? lo1 : lo0;
        u.h[1] = p;
        An[nt >> 1][mt] = u.v;
      }
    }
  }
}

__device__ __forceinline__ void copyA(v16h (&A)[6][2], const v16h (&An)[4][2]) {
#pragma unroll
  for (int kt = 0; kt < 4; ++kt) {
    A[kt][0] = An[kt][0];
    A[kt][1] = An[kt][1];
  }
}

// ---------------- Prep kernel: build transposed/reordered f16 weight image in d_ws ----------------
__global__ void prep_weights(const float* __restrict__ d1w0, const float* __restrict__ d1w1,
                             const float* __restrict__ d1w2, const float* __restrict__ d2w0,
                             const float* __restrict__ d2w1, const float* __restrict__ d2w2,
                             const float* __restrict__ d2w3, const float* __restrict__ cw0,
                             const float* __restrict__ cw1, h16* __restrict__ ws) {
  int i = blockIdx.x * blockDim.x + threadIdx.x;
  if (i >= LDS_HALFS) return;
  float v = 0.0f;
  if (i < OFF_D1W1)      { int j = i - OFF_D1W0; int n = j >> 6, k = j & 63;
                           v = (k < 63) ? d1w0[k*128 + n] : 0.0f; }
  else if (i < OFF_D1W2) { int j = i - OFF_D1W1; int n = j >> 7, k = j & 127; v = d1w1[k*128 + n]; }
  else if (i < OFF_D2W0) { int j = i - OFF_D1W2; int n = j >> 7, k = j & 127; v = d1w2[k*128 + n]; }
  else if (i < OFF_D2W1) { int j = i - OFF_D2W0; int n = j / 192, k = j % 192;
                           if (k < 128)      v = d2w0[(63 + k)*128 + n];
                           else if (k < 191) v = d2w0[(k - 128)*128 + n];
                           else              v = 0.0f; }
  else if (i < OFF_D2W2) { int j = i - OFF_D2W1; int n = j >> 7, k = j & 127; v = d2w1[k*128 + n]; }
  else if (i < OFF_D2W3) { int j = i - OFF_D2W2; int n = j >> 7, k = j & 127; v = d2w2[k*128 + n]; }
  else if (i < OFF_CW0)  { int j = i - OFF_D2W3; int n = j >> 7, k = j & 127;
                           v = (n < 128) ? d2w3[k*129 + 1 + n]
                             : ((n == 128) ? d2w3[k*129] : 0.0f); }
  else if (i < OFF_CW1)  { int j = i - OFF_CW0; int n = j / 160, k = j % 160;
                           if (k < 128)      v = cw0[(27 + k)*128 + n];
                           else if (k < 155) v = cw0[(k - 128)*128 + n];
                           else              v = 0.0f; }
  else                   { int j = i - OFF_CW1; int n = j >> 7, k = j & 127;
                           v = (n < 3) ? cw1[k*3 + n] : 0.0f; }
  ws[i] = (h16)v;
}

// ---------------- Main kernel ----------------
__global__ void __launch_bounds__(256, 1)
fused_nerf_wmma(const h16* __restrict__ wts, const float* __restrict__ x,
                float* __restrict__ out, int nrows, int tiles_total) {
  __shared__ __align__(16) h16 smem[LDS_HALFS];
  const int tid  = threadIdx.x;
  const int lane = tid & 31;
  const int wave = tid >> 5;
  const int m0 = lane & 15, hf = lane >> 4;

  // ---- Pull the weight image into LDS: TDM if available, else vectorized copy ----
#if __has_builtin(__builtin_amdgcn_tensor_load_to_lds)
  if (wave == 0) {
    // D# group0: count=1 | lds_addr | global_addr(57b) | type=2
    u32x4 g0;
    unsigned int ldsa = (unsigned int)(uintptr_t)&smem[0];
    unsigned long long ga = (unsigned long long)(uintptr_t)wts;
    g0[0] = 1u;
    g0[1] = ldsa;
    g0[2] = (unsigned int)ga;
    g0[3] = (unsigned int)((ga >> 32) & 0x1FFFFFFu) | (2u << 30);
    // D# group1: data_size=3 (8B); 2-D tile 2176 x 16 units; dim0 stride 2176
    i32x8 g1;
    g1[0] = (int)(3u << 16);        // workgroup_mask=0 | data_size=3
    g1[1] = (int)(2176u << 16);     // tensor_dim0[15:0] in bits 63:48
    g1[2] = (int)(16u << 16);       // tensor_dim0 hi=0 | tensor_dim1 lo=16
    g1[3] = (int)(2176u << 16);     // tensor_dim1 hi=0 | tile_dim0=2176
    g1[4] = 16;                     // tile_dim1=16 | tile_dim2=0
    g1[5] = 2176;                   // tensor_dim0_stride lo
    g1[6] = 0;
    g1[7] = 0;
    i32x4 z4 = {};
#if defined(__clang_major__) && __clang_major__ >= 23
    i32x8 z8 = {};
    __builtin_amdgcn_tensor_load_to_lds(g0, g1, z4, z4, z8, 0);
#else
    __builtin_amdgcn_tensor_load_to_lds(g0, g1, z4, z4, 0);
#endif
#if __has_builtin(__builtin_amdgcn_s_wait_tensorcnt)
    __builtin_amdgcn_s_wait_tensorcnt(0);
#else
    asm volatile("s_wait_tensorcnt 0x0" ::: "memory");
#endif
  }
#else
  {
    const v8h* src = (const v8h*)wts;
    v8h* dst = (v8h*)smem;
    for (int i = tid; i < LDS_HALFS / 8; i += 256) dst[i] = src[i];
  }
#endif
  __syncthreads();
  const h16* W = smem;

  // Identity B fragment for the register transpose: B[k][n] = (k==n), K=32, N=16.
  v16h idB;
  {
    union { v16h v; h16 e[16]; } f;
#pragma unroll
    for (int e = 0; e < 16; ++e) f.e[e] = (h16)(((e >> 3)*16 + hf*8 + (e & 7) == m0) ? 1.0f : 0.0f);
    idB = f.v;
  }

  // Double-buffered transpose operands; high halves stay zero for the whole kernel.
  TA ta[2];
  {
    v8h z = {};
    ta[0].h[0] = z; ta[0].h[1] = z;
    ta[1].h[0] = z; ta[1].h[1] = z;
  }

  // ---------------- Persistent loop over 256-row tiles (32 rows / wave) ----------------
  for (int t = blockIdx.x; t < tiles_total; t += gridDim.x) {
    const int row0 = t*256 + wave*32;
    if (row0 >= nrows) continue;

    // Input fragments straight from global into registers (reused by concat layers).
    v16h posA[2][2], viewA[2];
    if (row0 + 32 <= nrows) {       // full tile: compile-time masking, no exec juggling
#pragma unroll
      for (int mt = 0; mt < 2; ++mt) {
        const int row = row0 + mt*16 + m0;
        posA[0][mt] = load_x_frag_fast<0, 63>(x, row, 0,  hf);
        posA[1][mt] = load_x_frag_fast<0, 63>(x, row, 32, hf);
        viewA[mt]   = load_x_frag_fast<63, 27>(x, row, 0, hf);
      }
    } else {                        // tail tile: runtime masked
#pragma unroll
      for (int mt = 0; mt < 2; ++mt) {
        const int row = row0 + mt*16 + m0;
        posA[0][mt] = load_x_frag_slow(x, row, nrows, 0, 63, 0,  hf);
        posA[1][mt] = load_x_frag_slow(x, row, nrows, 0, 63, 32, hf);
        viewA[mt]   = load_x_frag_slow(x, row, nrows, 63, 27, 0, hf);
      }
    }

    v16h A[6][2], An[4][2];

    // density MLP part 1
    A[0][0] = posA[0][0]; A[0][1] = posA[0][1];
    A[1][0] = posA[1][0]; A[1][1] = posA[1][1];
    layer_T<2,true>(A, W + OFF_D1W0,  64, An, idB, ta, lane);
    copyA(A, An);
    layer_T<4,true>(A, W + OFF_D1W1, 128, An, idB, ta, lane);
    copyA(A, An);
    layer_T<4,true>(A, W + OFF_D1W2, 128, An, idB, ta, lane);   // An = part1

    // density MLP part 2: input = [part1 | pos] (weights row-reordered to match)
    copyA(A, An);
    A[4][0] = posA[0][0]; A[4][1] = posA[0][1];
    A[5][0] = posA[1][0]; A[5][1] = posA[1][1];
    layer_T<6,true>(A, W + OFF_D2W0, 192, An, idB, ta, lane);
    copyA(A, An);
    layer_T<4,true>(A, W + OFF_D2W1, 128, An, idB, ta, lane);
    copyA(A, An);
    layer_T<4,true>(A, W + OFF_D2W2, 128, An, idB, ta, lane);

    // out2 = h @ d2_W3 (no relu): feature -> An; density from the SAME register A.
    copyA(A, An);
    layer_T<4,false>(A, W + OFF_D2W3, 128, An, idB, ta, lane);
    {
      v16h b[4];
#pragma unroll
      for (int kt = 0; kt < 4; ++kt) b[kt] = load_fragB(W + OFF_D2W3, 128, 128, kt*32, lane);
      v8f a0 = {}, a1 = {};
#pragma unroll
      for (int kt = 0; kt < 4; ++kt) {
        a0 = WMMA(A[kt][0], b[kt], a0);
        a1 = WMMA(A[kt][1], b[kt], a1);
      }
      if (m0 == 0) {
#pragma unroll
        for (int r = 0; r < 8; ++r) {
          int ra = row0 + r + 8*hf;
          int rb = row0 + 16 + r + 8*hf;
          if (ra < nrows) __builtin_nontemporal_store(a0[r], &out[(size_t)ra*4 + 3]);
          if (rb < nrows) __builtin_nontemporal_store(a1[r], &out[(size_t)rb*4 + 3]);
        }
      }
    }

    // color MLP: input = [feature | view] (weights row-reordered to match)
    copyA(A, An);
    A[4][0] = viewA[0]; A[4][1] = viewA[1];
    layer_T<5,true>(A, W + OFF_CW0, 160, An, idB, ta, lane);

    // rgb = hidden @ c_W1 -> out[...,0..2]
    {
      v16h b[4];
#pragma unroll
      for (int kt = 0; kt < 4; ++kt) b[kt] = load_fragB(W + OFF_CW1, 128, 0, kt*32, lane);
      v8f a0 = {}, a1 = {};
#pragma unroll
      for (int kt = 0; kt < 4; ++kt) {
        a0 = WMMA(An[kt][0], b[kt], a0);
        a1 = WMMA(An[kt][1], b[kt], a1);
      }
      if (m0 < 3) {
#pragma unroll
        for (int r = 0; r < 8; ++r) {
          int ra = row0 + r + 8*hf;
          int rb = row0 + 16 + r + 8*hf;
          if (ra < nrows) __builtin_nontemporal_store(a0[r], &out[(size_t)ra*4 + m0]);
          if (rb < nrows) __builtin_nontemporal_store(a1[r], &out[(size_t)rb*4 + m0]);
        }
      }
    }
  }
}

extern "C" void kernel_launch(void* const* d_in, const int* in_sizes, int n_in,
                              void* d_out, int out_size, void* d_ws, size_t ws_size,
                              hipStream_t stream) {
  const float* x    = (const float*)d_in[0];
  const float* d1w0 = (const float*)d_in[1];
  const float* d1w1 = (const float*)d_in[2];
  const float* d1w2 = (const float*)d_in[3];
  const float* d2w0 = (const float*)d_in[4];
  const float* d2w1 = (const float*)d_in[5];
  const float* d2w2 = (const float*)d_in[6];
  const float* d2w3 = (const float*)d_in[7];
  const float* cw0  = (const float*)d_in[8];
  const float* cw1  = (const float*)d_in[9];
  float* out = (float*)d_out;
  h16* ws = (h16*)d_ws;   // 278,528 B weight image

  // Build the f16 weight image in workspace (graph-capture safe: plain kernel launch).
  prep_weights<<<dim3((LDS_HALFS + 255) / 256), dim3(256), 0, stream>>>(
      d1w0, d1w1, d1w2, d2w0, d2w1, d2w2, d2w3, cw0, cw1, ws);

  const int nrows = in_sizes[0] / XCH;
  const int tiles = (nrows + 255) / 256;
  int grid = tiles < 1024 ? tiles : 1024;   // persistent blocks

  fused_nerf_wmma<<<dim3(grid), dim3(256), 0, stream>>>(ws, x, out, nrows, tiles);
}
